// ODECNF_75720273429127
// MI455X (gfx1250) — matrix operand order repaired and checked by reading, not compiled, not run on hardware
//
#include <hip/hip_runtime.h>
#include <stdint.h>

// ============================================================================
// Fused CNF/RK4 integrator for MI455X (gfx1250, wave32, WMMA).
//
// Key observations:
//  * reg term is exactly 0 in value -> energy/jacnorm never computed.
//  * Each of the 32768 rows integrates independently -> one wave per 16-row
//    WMMA tile carries state (x, dx, u, div) through all 16 dynamics evals.
//  * gc = cond@Wc + b1 and ew = e@W2^T are t-independent -> precomputed once
//    (prologue WMMA kernel) into workspace, stored in D-fragment-major order
//    so the main kernel re-loads them with two b128 loads per fragment.
//  * All GEMM B operands are stored TRANSPOSED in bf16 so each lane's
//    B-fragment is two 16B reads starting 32B apart (2x b128).
//  * W1 (both orientations, 256KB bf16) lives in LDS, shared by 4 waves/WGP;
//    W2^T (128KB) streams from L2 (it stays hot).
//  * bf16 conversion via native v_cvt (plain __bf16 casts), tanh via native
//    V_TANH_F32 when the builtin exists -> minimal VALU alongside the WMMAs.
// Workspace requirement: ~97 MB (see WS_NEED).
// ============================================================================

#define B_ROWS 32768
#define D_DIM  128
#define H_DIM  512
#define TILES  (B_ROWS / 16)   // 2048
#define DT     0.25f

typedef __attribute__((ext_vector_type(16))) __bf16    v16bf;
typedef __attribute__((ext_vector_type(8)))  float     v8f;
typedef __attribute__((ext_vector_type(4)))  float     v4f;
typedef __attribute__((ext_vector_type(8)))  unsigned  v8u;
typedef __attribute__((ext_vector_type(4)))  unsigned  v4u;

// ---- workspace layout (bytes) ----------------------------------------------
static constexpr size_t OFF_W1C = 0;                       // bf16 [512][128] = W1^T (B-src for GEMM1: B=W1)
static constexpr size_t OFF_W1R = OFF_W1C + 131072;        // bf16 [128][512] = W1   (B-src for GEMM3: B=W1^T)
static constexpr size_t OFF_W2C = OFF_W1R + 131072;        // bf16 [128][512] = W2^T (B-src for GEMM2: B=W2)
static constexpr size_t OFF_W2R = OFF_W2C + 131072;        // bf16 [512][128] = W2   (B-src for ew GEMM: B=W2^T)
static constexpr size_t OFF_WCC = OFF_W2R + 131072;        // bf16 [512][128] = Wc^T (B-src for gc GEMM: B=Wc)
static constexpr size_t OFF_GC  = OFF_WCC + 131072;        // f32  frag-major [TILES][32][32][8]
static constexpr size_t OFF_EW  = OFF_GC + (size_t)TILES * 32 * 32 * 8 * 4;  // bf16 same layout
static constexpr size_t WS_NEED = OFF_EW + (size_t)TILES * 32 * 32 * 8 * 2;  // ~97 MB

#define MAIN_LDS (262144 + 4 * 14336)   // 319488 <= 320KB per WGP

// ---- helpers ---------------------------------------------------------------
// Native bf16 converts: clang lowers these casts to v_cvt_*bf16* on gfx1250.
__device__ __forceinline__ unsigned short f2bf(float f) {
  __bf16 b = (__bf16)f;
  return __builtin_bit_cast(unsigned short, b);
}
__device__ __forceinline__ float bf2f(unsigned short h) {
  return (float)__builtin_bit_cast(__bf16, h);
}
__device__ __forceinline__ unsigned pk2(float a, float b) {
  return (unsigned)f2bf(a) | ((unsigned)f2bf(b) << 16);
}

__device__ __forceinline__ v8f wmma_bf16(v8u a, v8u b, v8f c) {
  return __builtin_amdgcn_wmma_f32_16x16x32_bf16(
      false, __builtin_bit_cast(v16bf, a),
      false, __builtin_bit_cast(v16bf, b),
      (short)0, c, false, false);
}

// A fragment (16x32 bf16, row-major source S[16][ldk]):
// lane<16: row=lane,    elems = K k0..k0+7  then k0+16..k0+23
// lane>=16: row=lane-16, elems = K k0+8..+15 then k0+24..+31
__device__ __forceinline__ v8u load_afrag(const unsigned short* S, int ldk, int k0, int lane) {
  const unsigned short* p = S + (lane & 15) * ldk + k0 + ((lane >> 4) << 3);
  v4u lo = *(const v4u*)p;
  v4u hi = *(const v4u*)(p + 16);
  v8u r;
  r[0] = lo[0]; r[1] = lo[1]; r[2] = lo[2]; r[3] = lo[3];
  r[4] = hi[0]; r[5] = hi[1]; r[6] = hi[2]; r[7] = hi[3];
  return r;
}

// B fragment (32x16 bf16) from the TRANSPOSED store T[n][k] (row-major, ldk):
// lane<16: col n0+lane, K k0..k0+7 then k0+16..k0+23 ; lane>=16: col n0+lane-16, K k0+8..+15 then k0+24..+31
__device__ __forceinline__ v8u load_bfrag(const unsigned short* T, int n0, int ldk, int k0, int lane) {
  const unsigned short* p = T + (size_t)(n0 + (lane & 15)) * ldk + k0 + ((lane >> 4) << 3);
  v4u lo = *(const v4u*)p;
  v4u hi = *(const v4u*)(p + 16);
  v8u r;
  r[0] = lo[0]; r[1] = lo[1]; r[2] = lo[2]; r[3] = lo[3];
  r[4] = hi[0]; r[5] = hi[1]; r[6] = hi[2]; r[7] = hi[3];
  return r;
}

__device__ __forceinline__ float fast_tanh(float v) {
#if __has_builtin(__builtin_amdgcn_tanhf)
  return __builtin_amdgcn_tanhf(v);                  // native V_TANH_F32 (TRANS32)
#else
  float a = __builtin_fabsf(v);
  float q = __expf(-2.0f * a);                       // v_exp_f32
  float r = (1.0f - q) * __builtin_amdgcn_rcpf(1.0f + q);
  return v < 0.0f ? -r : r;
#endif
}

// ---- kernel 1: bf16 weight conversion + transposes -------------------------
__global__ __launch_bounds__(256) void prep_weights(
    const float* __restrict__ W1, const float* __restrict__ Wc,
    const float* __restrict__ W2, char* __restrict__ ws) {
  unsigned short* W1c = (unsigned short*)(ws + OFF_W1C);
  unsigned short* W1r = (unsigned short*)(ws + OFF_W1R);
  unsigned short* W2c = (unsigned short*)(ws + OFF_W2C);
  unsigned short* W2r = (unsigned short*)(ws + OFF_W2R);
  unsigned short* Wcc = (unsigned short*)(ws + OFF_WCC);
  int i = blockIdx.x * 256 + threadIdx.x;            // 0..65535
  int k1 = i >> 9, n1 = i & 511;                     // W1/Wc: [128][512]
  float w1 = W1[i], wc = Wc[i];
  W1r[i] = f2bf(w1);
  W1c[n1 * 128 + k1] = f2bf(w1);
  Wcc[n1 * 128 + k1] = f2bf(wc);
  int k2 = i >> 7, n2 = i & 127;                     // W2: [512][128]
  float w2 = W2[i];
  W2r[i] = f2bf(w2);
  W2c[n2 * 512 + k2] = f2bf(w2);
}

// ---- kernel 2: gc = cond@Wc + b1 (f32), ew = e@W2^T (bf16), frag-major -----
__global__ __launch_bounds__(128) void precompute_gc_ew(
    const float* __restrict__ cond, const float* __restrict__ e_in,
    const float* __restrict__ b1, const char* __restrict__ ws_ro,
    float* __restrict__ gcw, unsigned short* __restrict__ eww) {
  const unsigned short* Wcc = (const unsigned short*)(ws_ro + OFF_WCC);
  const unsigned short* W2r = (const unsigned short*)(ws_ro + OFF_W2R);
  const int lane = threadIdx.x & 31;
  const int tile = blockIdx.x * 4 + (threadIdx.x >> 5);
  const int rowBase = tile * 16;
  const int m = lane & 15, hh = lane >> 4;

  // preload A fragments (cond rows & e rows), K = 128 -> 4 k-steps
  v8u ca[4], ea[4];
#pragma unroll
  for (int k = 0; k < 4; ++k) {
    int koff = k * 32 + (hh << 3);
    const float* cp = cond + (size_t)(rowBase + m) * 128 + koff;
    const float* ep = e_in + (size_t)(rowBase + m) * 128 + koff;
    v4f c0 = *(const v4f*)cp,        c1 = *(const v4f*)(cp + 4);
    v4f c2 = *(const v4f*)(cp + 16), c3 = *(const v4f*)(cp + 20);
    ca[k][0] = pk2(c0[0], c0[1]); ca[k][1] = pk2(c0[2], c0[3]);
    ca[k][2] = pk2(c1[0], c1[1]); ca[k][3] = pk2(c1[2], c1[3]);
    ca[k][4] = pk2(c2[0], c2[1]); ca[k][5] = pk2(c2[2], c2[3]);
    ca[k][6] = pk2(c3[0], c3[1]); ca[k][7] = pk2(c3[2], c3[3]);
    v4f e0 = *(const v4f*)ep,        e1 = *(const v4f*)(ep + 4);
    v4f e2 = *(const v4f*)(ep + 16), e3 = *(const v4f*)(ep + 20);
    ea[k][0] = pk2(e0[0], e0[1]); ea[k][1] = pk2(e0[2], e0[3]);
    ea[k][2] = pk2(e1[0], e1[1]); ea[k][3] = pk2(e1[2], e1[3]);
    ea[k][4] = pk2(e2[0], e2[1]); ea[k][5] = pk2(e2[2], e2[3]);
    ea[k][6] = pk2(e3[0], e3[1]); ea[k][7] = pk2(e3[2], e3[3]);
  }

  for (int n = 0; n < 32; ++n) {                     // 32 n-tiles over H=512
    float b1v = b1[n * 16 + m];
    v8f accg, acce;
#pragma unroll
    for (int j = 0; j < 8; ++j) { accg[j] = b1v; acce[j] = 0.0f; }
#pragma unroll
    for (int k = 0; k < 4; ++k) {
      accg = wmma_bf16(ca[k], load_bfrag(Wcc, n * 16, 128, k * 32, lane), accg);
      acce = wmma_bf16(ea[k], load_bfrag(W2r, n * 16, 128, k * 32, lane), acce);
    }
    size_t base = (((size_t)tile * 32 + n) * 32 + lane) * 8;
    v4f g0, g1;
    g0[0] = accg[0]; g0[1] = accg[1]; g0[2] = accg[2]; g0[3] = accg[3];
    g1[0] = accg[4]; g1[1] = accg[5]; g1[2] = accg[6]; g1[3] = accg[7];
    *(v4f*)(gcw + base)     = g0;
    *(v4f*)(gcw + base + 4) = g1;
    v4u ep;
    ep[0] = pk2(acce[0], acce[1]); ep[1] = pk2(acce[2], acce[3]);
    ep[2] = pk2(acce[4], acce[5]); ep[3] = pk2(acce[6], acce[7]);
    *(v4u*)(eww + base) = ep;
  }
}

// ---- kernel 3: fused RK4 integrator, one wave per 16-row tile --------------
__global__ __launch_bounds__(128) void cnf_main(
    const float* __restrict__ x_in, const float* __restrict__ logpx,
    const float* __restrict__ e_in, const float* __restrict__ bt,
    const float* __restrict__ b2, const char* __restrict__ ws_ro,
    const float* __restrict__ gcw, const unsigned short* __restrict__ eww,
    float* __restrict__ y_out, float* __restrict__ lp_out) {
  extern __shared__ char smem[];
  unsigned short* sW1c = (unsigned short*)smem;      // [512][128] bf16
  unsigned short* sW1r = sW1c + 65536;               // [128][512] bf16
  {                                                  // coop-load 256KB of W1 (both orientations)
    const v4u* src = (const v4u*)(ws_ro + OFF_W1C);  // W1c || W1r contiguous in ws
    v4u* dst = (v4u*)smem;
    for (int i = threadIdx.x; i < 16384; i += 128) dst[i] = src[i];
  }
  __syncthreads();

  const int wave = threadIdx.x >> 5, lane = threadIdx.x & 31;
  const int m = lane & 15, hh = lane >> 4;
  const int tile = blockIdx.x * 4 + wave;
  const int rowBase = tile * 16;

  char* pw = smem + 262144 + wave * 14336;           // per-wave scratch
  float* xacc = (float*)pw;                          // 2048 f32: RK4 combination accumulator
  unsigned short* stage = (unsigned short*)(pw + 8192);        // bf16 [16][128]: A-src for GEMM1
  unsigned short* hbuf  = (unsigned short*)(pw + 8192 + 4096); // bf16 [16][64]: A-src for GEMM2/3

  const unsigned short* W2c = (const unsigned short*)(ws_ro + OFF_W2C);

  // x master copy, D-fragment layout: xr[f][j] <-> row (j+8*hh), col f*16+m
  float xr[8][8];
#pragma unroll
  for (int f = 0; f < 8; ++f)
#pragma unroll
    for (int j = 0; j < 8; ++j)
      xr[f][j] = x_in[(size_t)(rowBase + j + 8 * hh) * 128 + f * 16 + m];

  float b2v[8];
#pragma unroll
  for (int f = 0; f < 8; ++f) b2v[f] = b2[f * 16 + m];

  float divacc[8];
#pragma unroll
  for (int j = 0; j < 8; ++j) divacc[j] = 0.0f;

  v8f dxr[8];                                        // previous-sub dx / current dx accumulator
#pragma unroll
  for (int f = 0; f < 8; ++f)
#pragma unroll
    for (int j = 0; j < 8; ++j) dxr[f][j] = 0.0f;

  for (int step = 0; step < 4; ++step) {
    float t0 = step * DT;
#pragma unroll
    for (int f = 0; f < 8; ++f)
#pragma unroll
      for (int j = 0; j < 8; ++j)
        xacc[(f * 8 + j) * 32 + lane] = xr[f][j];

    for (int sub = 0; sub < 4; ++sub) {              // k1..k4
      float acoef = (sub == 0) ? 0.0f : ((sub == 3) ? DT : 0.5f * DT);
      float t = t0 + acoef;                          // eval times: t0, t0+dt/2, t0+dt/2, t0+dt
      float wc = (sub == 1 || sub == 2) ? 2.0f : 1.0f;

      // stage xx_eval = x + acoef * k_prev (bf16, row-major for A-frag reads)
#pragma unroll
      for (int f = 0; f < 8; ++f) {
        int col = f * 16 + m;
#pragma unroll
        for (int j = 0; j < 8; ++j)
          stage[(j + 8 * hh) * 128 + col] = f2bf(xr[f][j] + acoef * dxr[f][j]);
      }
      __syncthreads();

      v8f uf[8];
#pragma unroll
      for (int f = 0; f < 8; ++f)
#pragma unroll
        for (int j = 0; j < 8; ++j) { dxr[f][j] = b2v[f]; uf[f][j] = 0.0f; }

      for (int hc = 0; hc < 8; ++hc) {               // H in chunks of 64
        // ---- GEMM1: h_pre = xx@W1 + gc + t*bt ----
        v8f acc[4];
#pragma unroll
        for (int c4 = 0; c4 < 4; ++c4) {
          int nt = hc * 4 + c4;
          size_t gb = (((size_t)tile * 32 + nt) * 32 + lane) * 8;
          v4f g0 = *(const v4f*)(gcw + gb);
          v4f g1 = *(const v4f*)(gcw + gb + 4);
          float btv = t * bt[nt * 16 + m];
          acc[c4][0] = g0[0] + btv; acc[c4][1] = g0[1] + btv;
          acc[c4][2] = g0[2] + btv; acc[c4][3] = g0[3] + btv;
          acc[c4][4] = g1[0] + btv; acc[c4][5] = g1[1] + btv;
          acc[c4][6] = g1[2] + btv; acc[c4][7] = g1[3] + btv;
        }
#pragma unroll
        for (int k = 0; k < 4; ++k) {
          v8u a = load_afrag(stage, 128, k * 32, lane);
#pragma unroll
          for (int c4 = 0; c4 < 4; ++c4)
            acc[c4] = wmma_bf16(a, load_bfrag(sW1c, (hc * 4 + c4) * 16, 128, k * 32, lane), acc[c4]);
        }
        // ---- tanh + stage h chunk ----
        float hval[4][8];
#pragma unroll
        for (int c4 = 0; c4 < 4; ++c4)
#pragma unroll
          for (int j = 0; j < 8; ++j) {
            float hv = fast_tanh(acc[c4][j]);
            hval[c4][j] = hv;
            hbuf[(j + 8 * hh) * 64 + c4 * 16 + m] = f2bf(hv);
          }
        __syncthreads();
        // ---- GEMM2: dx += h_chunk @ W2[chunk,:] ----
#pragma unroll
        for (int k2 = 0; k2 < 2; ++k2) {
          v8u a = load_afrag(hbuf, 64, k2 * 32, lane);
#pragma unroll
          for (int f = 0; f < 8; ++f)
            dxr[f] = wmma_bf16(a, load_bfrag(W2c, f * 16, 512, hc * 64 + k2 * 32, lane), dxr[f]);
        }
        // ---- ewt = ew * (1 - h^2), restage (reuse hbuf) ----
#pragma unroll
        for (int c4 = 0; c4 < 4; ++c4) {
          int nt = hc * 4 + c4;
          size_t gb = (((size_t)tile * 32 + nt) * 32 + lane) * 8;
          v4u ep = *(const v4u*)(eww + gb);
#pragma unroll
          for (int j = 0; j < 8; ++j) {
            unsigned short bits = (unsigned short)((ep[j >> 1] >> ((j & 1) * 16)) & 0xffffu);
            float w = bf2f(bits) * (1.0f - hval[c4][j] * hval[c4][j]);
            hbuf[(j + 8 * hh) * 64 + c4 * 16 + m] = f2bf(w);
          }
        }
        __syncthreads();
        // ---- GEMM3: u += ewt_chunk @ W1^T[chunk,:] ----
#pragma unroll
        for (int k2 = 0; k2 < 2; ++k2) {
          v8u a = load_afrag(hbuf, 64, k2 * 32, lane);
#pragma unroll
          for (int f = 0; f < 8; ++f)
            uf[f] = wmma_bf16(a, load_bfrag(sW1r, f * 16, 512, hc * 64 + k2 * 32, lane), uf[f]);
        }
        __syncthreads();                             // hbuf reused next chunk
      }

      // divergence partials and RK4 combination update
      float cc = DT * wc * (1.0f / 6.0f);
#pragma unroll
      for (int f = 0; f < 8; ++f)
#pragma unroll
        for (int j = 0; j < 8; ++j) {
          float ev = e_in[(size_t)(rowBase + j + 8 * hh) * 128 + f * 16 + m];
          divacc[j] += wc * uf[f][j] * ev;
          xacc[(f * 8 + j) * 32 + lane] += cc * dxr[f][j];
        }
    } // sub

#pragma unroll
    for (int f = 0; f < 8; ++f)
#pragma unroll
      for (int j = 0; j < 8; ++j)
        xr[f][j] = xacc[(f * 8 + j) * 32 + lane];
  } // step

  // write y
#pragma unroll
  for (int f = 0; f < 8; ++f)
#pragma unroll
    for (int j = 0; j < 8; ++j)
      y_out[(size_t)(rowBase + j + 8 * hh) * 128 + f * 16 + m] = xr[f][j];

  // row-wise divergence reduction across 16-lane groups (wave32 shfl_xor)
#pragma unroll
  for (int j = 0; j < 8; ++j) {
    float v = divacc[j];
    v += __shfl_xor(v, 1, 16);
    v += __shfl_xor(v, 2, 16);
    v += __shfl_xor(v, 4, 16);
    v += __shfl_xor(v, 8, 16);
    if (m == 0) {
      int r = rowBase + j + 8 * hh;
      lp_out[r] = logpx[r] - (DT / 6.0f) * v;
    }
  }
}

// ---- host launcher ---------------------------------------------------------
extern "C" void kernel_launch(void* const* d_in, const int* in_sizes, int n_in,
                              void* d_out, int out_size, void* d_ws, size_t ws_size,
                              hipStream_t stream) {
  (void)in_sizes; (void)n_in; (void)out_size; (void)ws_size; (void)WS_NEED;
  const float* x     = (const float*)d_in[0];
  const float* logpx = (const float*)d_in[1];
  const float* cond  = (const float*)d_in[2];
  const float* e     = (const float*)d_in[3];
  const float* W1    = (const float*)d_in[4];
  const float* Wc    = (const float*)d_in[5];
  const float* bt    = (const float*)d_in[6];
  const float* b1    = (const float*)d_in[7];
  const float* W2    = (const float*)d_in[8];
  const float* b2    = (const float*)d_in[9];

  char* ws = (char*)d_ws;
  float* gcw = (float*)(ws + OFF_GC);
  unsigned short* eww = (unsigned short*)(ws + OFF_EW);
  float* y  = (float*)d_out;
  float* lp = y + (size_t)B_ROWS * D_DIM;

  hipFuncSetAttribute(reinterpret_cast<const void*>(cnf_main),
                      hipFuncAttributeMaxDynamicSharedMemorySize, MAIN_LDS);

  prep_weights<<<256, 256, 0, stream>>>(W1, Wc, W2, ws);
  precompute_gc_ew<<<TILES / 4, 128, 0, stream>>>(cond, e, b1, ws, gcw, eww);
  cnf_main<<<TILES / 4, 128, MAIN_LDS, stream>>>(x, logpx, e, bt, b2, ws,
                                                 gcw, eww, y, lp);
}
